// EGNNNetwork_90752658965038
// MI455X (gfx1250) — matrix operand order repaired and checked by reading, not compiled, not run on hardware
//
#include <hip/hip_runtime.h>

typedef __attribute__((ext_vector_type(16))) _Float16 v16h;
typedef __attribute__((ext_vector_type(8)))  float    v8f;
typedef __attribute__((ext_vector_type(4)))  unsigned int u32x4;

#define N_NODES_C  50000
#define N_EDGES_C  800000
#define EPRIME_C   850000
#define N_GRAPHS_C 64
#define NF_C       110        // NF_EMB
#define MDIM_C     32
#define EHID_C     128
#define NHID_C     128
#define EF_K_C     222
#define EF_KP_C    224        // padded K for edge GEMM1
#define NODE_K_C   142
#define NODE_KP_C  160        // padded K for node GEMM1
#define NODE_NP_C  112        // padded N for node GEMM2 (110 -> 112)

union FragU { u32x4 q[2]; v16h v; };

// A fragment (16x32 f16): lane m = lane&15; halves e=0..7 -> K = hl+e,
// e=8..15 -> K = 16+hl+(e-8), hl = (lane<16)?0:8. Two contiguous 16B runs.
__device__ __forceinline__ v16h load_a16(const _Float16* row, int kc, int hl) {
    FragU u;
    u.q[0] = *(const u32x4*)(row + kc * 32 + hl);
    u.q[1] = *(const u32x4*)(row + kc * 32 + 16 + hl);
    return u.v;
}

// B fragment from pre-swizzled weights: 32 lanes x 16 halves contiguous per tile.
__device__ __forceinline__ v16h load_b16(const _Float16* w, int tile, int lane) {
    FragU u;
    const u32x4* p = (const u32x4*)(w + ((size_t)tile * 32 + lane) * 16);
    u.q[0] = p[0]; u.q[1] = p[1];
    return u.v;
}

__device__ __forceinline__ float siluf(float x) { return x / (1.0f + __expf(-x)); }

// ---------------------------------------------------------------------------
// Weight swizzle: W[K,N] fp32 row-major -> per-(kc,nt) tile of 32 lanes x 16
// halves. lane: n = nt*16 + (lane&15); hl = (lane<16?0:8);
// k = kc*32 + hl + e for e<8, kc*32 + 16 + hl + (e-8) for e>=8.
// ---------------------------------------------------------------------------
__global__ void swizzle_w_kernel(const float* __restrict__ W, int K, int N,
                                 int ntCount, _Float16* __restrict__ out) {
    int t    = blockIdx.x;
    int kc   = t / ntCount;
    int nt   = t % ntCount;
    int lane = threadIdx.x;
    int n    = nt * 16 + (lane & 15);
    int hl   = (lane < 16) ? 0 : 8;
    _Float16* dst = out + ((size_t)t * 32 + lane) * 16;
    for (int e = 0; e < 16; ++e) {
        int k = kc * 32 + (e < 8 ? hl + e : 16 + hl + (e - 8));
        float v = (k < K && n < N) ? W[(size_t)k * N + n] : 0.0f;
        dst[e] = (_Float16)v;
    }
}

// ---------------------------------------------------------------------------
// Embedding: feats[n, 0:6] = x[n, {0,11,12,13,14,15}], then 10 lookups.
// ---------------------------------------------------------------------------
__global__ __launch_bounds__(256) void embed_kernel(
    const float* __restrict__ x,
    const float* e0, const float* e1, const float* e2, const float* e3,
    const float* e4, const float* e5, const float* e6, const float* e7,
    const float* e8, const float* e9,
    float* __restrict__ feats, _Float16* __restrict__ feats16) {
    int n = blockIdx.x * 256 + threadIdx.x;
    if (n >= N_NODES_C) return;
    const float* xr = x + (size_t)n * 16;
    float* fr = feats + (size_t)n * NF_C;
    _Float16* hr = feats16 + (size_t)n * NF_C;
    const float* tabs[10] = {e0,e1,e2,e3,e4,e5,e6,e7,e8,e9};
    const int dims[10]    = {32,8,8,8,8,8,8,8,8,8};
    float v;
    v = xr[0];  fr[0]=v; hr[0]=(_Float16)v;
    v = xr[11]; fr[1]=v; hr[1]=(_Float16)v;
    v = xr[12]; fr[2]=v; hr[2]=(_Float16)v;
    v = xr[13]; fr[3]=v; hr[3]=(_Float16)v;
    v = xr[14]; fr[4]=v; hr[4]=(_Float16)v;
    v = xr[15]; fr[5]=v; hr[5]=(_Float16)v;
    int o = 6;
    for (int j = 0; j < 10; ++j) {
        int idx = (int)xr[1 + j];
        const float* t = tabs[j] + (size_t)idx * dims[j];
        for (int d = 0; d < dims[j]; ++d) {
            v = t[d]; fr[o + d] = v; hr[o + d] = (_Float16)v;
        }
        o += dims[j];
    }
}

// ---------------------------------------------------------------------------
// Per-edge attr: [edge_attr, ||pos_src - pos_dst||^2]  (loops: attr=0, rel=0)
// ---------------------------------------------------------------------------
__global__ __launch_bounds__(256) void ea_kernel(
    const int* __restrict__ ei, const float* __restrict__ eattr,
    const float* __restrict__ pos, _Float16* __restrict__ ea16) {
    int e = blockIdx.x * 256 + threadIdx.x;
    if (e >= EPRIME_C) return;
    int s, d; float a;
    if (e < N_EDGES_C) { s = ei[e]; d = ei[N_EDGES_C + e]; a = eattr[e]; }
    else               { s = d = e - N_EDGES_C; a = 0.0f; }
    float dx = pos[(size_t)s*3+0] - pos[(size_t)d*3+0];
    float dy = pos[(size_t)s*3+1] - pos[(size_t)d*3+1];
    float dz = pos[(size_t)s*3+2] - pos[(size_t)d*3+2];
    float rd = dx*dx + dy*dy + dz*dz;
    ea16[(size_t)e*2+0] = (_Float16)a;
    ea16[(size_t)e*2+1] = (_Float16)rd;
}

__global__ __launch_bounds__(256) void count_kernel(const int* __restrict__ batch,
                                                    float* __restrict__ counts) {
    int n = blockIdx.x * 256 + threadIdx.x;
    if (n < N_NODES_C) atomicAdd(&counts[batch[n]], 1.0f);
}

__global__ __launch_bounds__(256) void stats_kernel(const float* __restrict__ feats,
                                                    const int* __restrict__ batch,
                                                    float* __restrict__ gsum,
                                                    float* __restrict__ gsumsq) {
    int n = blockIdx.x * 256 + threadIdx.x;
    if (n >= N_NODES_C) return;
    const float* f = feats + (size_t)n * NF_C;
    float s = 0.0f, s2 = 0.0f;
    for (int c = 0; c < NF_C; ++c) { float v = f[c]; s += v; s2 += v * v; }
    int g = batch[n];
    atomicAdd(&gsum[g], s);
    atomicAdd(&gsumsq[g], s2);
}

__global__ void lnstat_fin_kernel(const float* counts, const float* gsum,
                                  const float* gsumsq, float* gmean, float* ginvstd) {
    int g = threadIdx.x;
    if (g >= N_GRAPHS_C) return;
    float nc  = fmaxf(counts[g], 1.0f);
    float den = nc * (float)NF_C;
    float m   = gsum[g] / den;
    float var = gsumsq[g] / den - m * m;
    gmean[g]   = m;
    ginvstd[g] = rsqrtf(var + 1e-5f);
}

// ---------------------------------------------------------------------------
// Edge MLP: per wave 16 edges. A = [feats16[dst] | feats16[src] | ea] (222->224)
// GEMM1: 16x224 @ 224x128 (7 kc x 8 nt WMMA) -> SiLU -> LDS
// GEMM2: 16x128 @ 128x32  (4 kc x 2 nt WMMA) -> SiLU -> atomic scatter to m_i
// ---------------------------------------------------------------------------
__global__ __launch_bounds__(128) void edge_mlp_kernel(
    const _Float16* __restrict__ feats16, const _Float16* __restrict__ ea16,
    const int* __restrict__ ei,
    const _Float16* __restrict__ w1s, const float* __restrict__ b1,
    const _Float16* __restrict__ w2s, const float* __restrict__ b2,
    float* __restrict__ m_i) {
    __shared__ __align__(16) _Float16 sA[4][16][232];   // stride 232 halves (16B mult)
    __shared__ __align__(16) _Float16 sH[4][16][136];
    int wave = threadIdx.x >> 5;
    int lane = threadIdx.x & 31;
    int baseE = blockIdx.x * 64 + wave * 16;

    // Stage A tile (f16) — cooperative gather by the wave's 32 lanes.
    for (int r = 0; r < 16; ++r) {
        int e = baseE + r;
        bool valid = e < EPRIME_C;
        int s = 0, d = 0;
        if (valid) {
            if (e < N_EDGES_C) { s = ei[e]; d = ei[N_EDGES_C + e]; }
            else               { s = d = e - N_EDGES_C; }
        }
        const _Float16* fd = feats16 + (size_t)d * NF_C;
        const _Float16* fs = feats16 + (size_t)s * NF_C;
        for (int c = lane; c < EF_KP_C; c += 32) {
            _Float16 v = (_Float16)0.0f;
            if (valid) {
                if      (c < NF_C)     v = fd[c];
                else if (c < 2*NF_C)   v = fs[c - NF_C];
                else if (c == 220)     v = ea16[(size_t)e*2+0];
                else if (c == 221)     v = ea16[(size_t)e*2+1];
            }
            sA[wave][r][c] = v;
        }
    }
    __syncthreads();

    int mrow = lane & 15;
    int hl   = (lane < 16) ? 0 : 8;
    int mout = (lane < 16) ? 0 : 8;
    int ncol = lane & 15;

    v8f acc[8];
    #pragma unroll
    for (int nt = 0; nt < 8; ++nt) acc[nt] = (v8f){0,0,0,0,0,0,0,0};
    for (int kc = 0; kc < 7; ++kc) {
        v16h a = load_a16(&sA[wave][mrow][0], kc, hl);
        #pragma unroll
        for (int nt = 0; nt < 8; ++nt) {
            v16h b = load_b16(w1s, kc * 8 + nt, lane);
            acc[nt] = __builtin_amdgcn_wmma_f32_16x16x32_f16(
                false, a, false, b, (short)0, acc[nt], false, false);
        }
    }
    // bias + SiLU, stage h1 as f16
    #pragma unroll
    for (int nt = 0; nt < 8; ++nt) {
        float bb = b1[nt * 16 + ncol];
        #pragma unroll
        for (int r = 0; r < 8; ++r) {
            float v = siluf(acc[nt][r] + bb);
            sH[wave][mout + r][nt * 16 + ncol] = (_Float16)v;
        }
    }
    __syncthreads();

    v8f acc2[2];
    acc2[0] = (v8f){0,0,0,0,0,0,0,0};
    acc2[1] = (v8f){0,0,0,0,0,0,0,0};
    for (int kc = 0; kc < 4; ++kc) {
        v16h a = load_a16(&sH[wave][mrow][0], kc, hl);
        #pragma unroll
        for (int nt = 0; nt < 2; ++nt) {
            v16h b = load_b16(w2s, kc * 2 + nt, lane);
            acc2[nt] = __builtin_amdgcn_wmma_f32_16x16x32_f16(
                false, a, false, b, (short)0, acc2[nt], false, false);
        }
    }
    // SiLU + scatter-add to m_i[dst]
    #pragma unroll
    for (int nt = 0; nt < 2; ++nt) {
        float bb = b2[nt * 16 + ncol];
        #pragma unroll
        for (int r = 0; r < 8; ++r) {
            int e = baseE + mout + r;
            if (e < EPRIME_C) {
                int d = (e < N_EDGES_C) ? ei[N_EDGES_C + e] : (e - N_EDGES_C);
                float v = siluf(acc2[nt][r] + bb);
                atomicAdd(&m_i[(size_t)d * MDIM_C + nt * 16 + ncol], v);
            }
        }
    }
}

// ---------------------------------------------------------------------------
// Node MLP: per wave 16 nodes. A = [LN(feats) | m_i] (142 -> 160)
// GEMM1: 16x160 @ 160x128 (5x8 WMMA) -> SiLU -> LDS
// GEMM2: 16x128 @ 128x112 (4x7 WMMA) -> residual into feats/feats16
// ---------------------------------------------------------------------------
__global__ __launch_bounds__(128) void node_mlp_kernel(
    float* __restrict__ feats, _Float16* __restrict__ feats16,
    const float* __restrict__ m_i, const int* __restrict__ batch,
    const float* __restrict__ gmean, const float* __restrict__ ginvstd,
    const float* __restrict__ lnw, const float* __restrict__ lnb,
    const _Float16* __restrict__ w1s, const float* __restrict__ b1,
    const _Float16* __restrict__ w2s, const float* __restrict__ b2) {
    __shared__ __align__(16) _Float16 sA[4][16][168];
    __shared__ __align__(16) _Float16 sH[4][16][136];
    int wave = threadIdx.x >> 5;
    int lane = threadIdx.x & 31;
    int baseN = blockIdx.x * 64 + wave * 16;

    for (int r = 0; r < 16; ++r) {
        int n = baseN + r;
        bool valid = n < N_NODES_C;
        int g = valid ? batch[n] : 0;
        float mu = gmean[g], is = ginvstd[g];
        const float* fr = feats + (size_t)n * NF_C;
        const float* mr = m_i  + (size_t)n * MDIM_C;
        for (int c = lane; c < NODE_KP_C; c += 32) {
            float v = 0.0f;
            if (valid) {
                if      (c < NF_C)     v = lnw[c] * ((fr[c] - mu) * is) + lnb[c];
                else if (c < NODE_K_C) v = mr[c - NF_C];
            }
            sA[wave][r][c] = (_Float16)v;
        }
    }
    __syncthreads();

    int mrow = lane & 15;
    int hl   = (lane < 16) ? 0 : 8;
    int mout = (lane < 16) ? 0 : 8;
    int ncol = lane & 15;

    v8f acc[8];
    #pragma unroll
    for (int nt = 0; nt < 8; ++nt) acc[nt] = (v8f){0,0,0,0,0,0,0,0};
    for (int kc = 0; kc < 5; ++kc) {
        v16h a = load_a16(&sA[wave][mrow][0], kc, hl);
        #pragma unroll
        for (int nt = 0; nt < 8; ++nt) {
            v16h b = load_b16(w1s, kc * 8 + nt, lane);
            acc[nt] = __builtin_amdgcn_wmma_f32_16x16x32_f16(
                false, a, false, b, (short)0, acc[nt], false, false);
        }
    }
    #pragma unroll
    for (int nt = 0; nt < 8; ++nt) {
        float bb = b1[nt * 16 + ncol];
        #pragma unroll
        for (int r = 0; r < 8; ++r) {
            float v = siluf(acc[nt][r] + bb);
            sH[wave][mout + r][nt * 16 + ncol] = (_Float16)v;
        }
    }
    __syncthreads();

    v8f acc2[7];
    #pragma unroll
    for (int nt = 0; nt < 7; ++nt) acc2[nt] = (v8f){0,0,0,0,0,0,0,0};
    for (int kc = 0; kc < 4; ++kc) {
        v16h a = load_a16(&sH[wave][mrow][0], kc, hl);
        #pragma unroll
        for (int nt = 0; nt < 7; ++nt) {
            v16h b = load_b16(w2s, kc * 7 + nt, lane);
            acc2[nt] = __builtin_amdgcn_wmma_f32_16x16x32_f16(
                false, a, false, b, (short)0, acc2[nt], false, false);
        }
    }
    // residual update
    #pragma unroll
    for (int nt = 0; nt < 7; ++nt) {
        int col = nt * 16 + ncol;
        float bb = (col < NF_C) ? b2[col] : 0.0f;
        #pragma unroll
        for (int r = 0; r < 8; ++r) {
            int n = baseN + mout + r;
            if (n < N_NODES_C && col < NF_C) {
                size_t idx = (size_t)n * NF_C + col;
                float v = feats[idx] + acc2[nt][r] + bb;
                feats[idx]   = v;
                feats16[idx] = (_Float16)v;
            }
        }
    }
}

__global__ __launch_bounds__(256) void pool_kernel(const float* __restrict__ feats,
                                                   const int* __restrict__ batch,
                                                   float* __restrict__ out) {
    int n = blockIdx.x * 256 + threadIdx.x;
    if (n >= N_NODES_C) return;
    int g = batch[n];
    const float* fr = feats + (size_t)n * NF_C;
    for (int c = 0; c < NF_C; ++c) atomicAdd(&out[(size_t)g * NF_C + c], fr[c]);
}

__global__ __launch_bounds__(256) void pool_div_kernel(float* __restrict__ out,
                                                       const float* __restrict__ counts) {
    int i = blockIdx.x * 256 + threadIdx.x;
    if (i >= N_GRAPHS_C * NF_C) return;
    out[i] /= fmaxf(counts[i / NF_C], 1.0f);
}

// ---------------------------------------------------------------------------
extern "C" void kernel_launch(void* const* d_in, const int* in_sizes, int n_in,
                              void* d_out, int out_size, void* d_ws, size_t ws_size,
                              hipStream_t stream) {
    (void)in_sizes; (void)n_in; (void)out_size; (void)ws_size;
    const float* x        = (const float*)d_in[0];
    const int*   ei       = (const int*)  d_in[1];
    const float* eattr    = (const float*)d_in[2];
    const float* pos      = (const float*)d_in[3];
    const int*   batch    = (const int*)  d_in[4];
    const float* embs[10];
    for (int j = 0; j < 10; ++j) embs[j] = (const float*)d_in[5 + j];
    const float* edge_w1  = (const float*)d_in[15];
    const float* edge_b1  = (const float*)d_in[16];
    const float* edge_w2  = (const float*)d_in[17];
    const float* edge_b2  = (const float*)d_in[18];
    const float* node_w1  = (const float*)d_in[19];
    const float* node_b1  = (const float*)d_in[20];
    const float* node_w2  = (const float*)d_in[21];
    const float* node_b2  = (const float*)d_in[22];
    const float* ln_w     = (const float*)d_in[23];
    const float* ln_b     = (const float*)d_in[24];
    float* out = (float*)d_out;

    // Workspace layout
    char* base = (char*)d_ws;
    size_t off = 0;
    auto alloc = [&](size_t bytes) -> char* {
        char* p = base + off;
        off = (off + bytes + 255) & ~(size_t)255;
        return p;
    };
    float*     feats   = (float*)    alloc((size_t)N_NODES_C * NF_C * 4);
    _Float16*  feats16 = (_Float16*) alloc((size_t)N_NODES_C * NF_C * 2);
    _Float16*  ea16    = (_Float16*) alloc((size_t)EPRIME_C * 2 * 2);
    float*     m_i     = (float*)    alloc((size_t)N_NODES_C * MDIM_C * 4);
    _Float16*  w1s     = (_Float16*) alloc((size_t)2 * EF_KP_C * EHID_C * 2);
    _Float16*  w2s     = (_Float16*) alloc((size_t)2 * EHID_C * MDIM_C * 2);
    _Float16*  nw1s    = (_Float16*) alloc((size_t)2 * NODE_KP_C * NHID_C * 2);
    _Float16*  nw2s    = (_Float16*) alloc((size_t)2 * NHID_C * NODE_NP_C * 2);
    float*     stats   = (float*)    alloc((size_t)5 * N_GRAPHS_C * 4);
    float* counts  = stats;
    float* gsum    = stats + 1 * N_GRAPHS_C;
    float* gsumsq  = stats + 2 * N_GRAPHS_C;
    float* gmean   = stats + 3 * N_GRAPHS_C;
    float* ginvstd = stats + 4 * N_GRAPHS_C;

    // Weight swizzle (per layer)
    for (int l = 0; l < 2; ++l) {
        swizzle_w_kernel<<<(EF_KP_C/32)*(EHID_C/16), 32, 0, stream>>>(
            edge_w1 + (size_t)l * EF_K_C * EHID_C, EF_K_C, EHID_C, EHID_C/16,
            w1s + (size_t)l * EF_KP_C * EHID_C);
        swizzle_w_kernel<<<(EHID_C/32)*(MDIM_C/16), 32, 0, stream>>>(
            edge_w2 + (size_t)l * EHID_C * MDIM_C, EHID_C, MDIM_C, MDIM_C/16,
            w2s + (size_t)l * EHID_C * MDIM_C);
        swizzle_w_kernel<<<(NODE_KP_C/32)*(NHID_C/16), 32, 0, stream>>>(
            node_w1 + (size_t)l * NODE_K_C * NHID_C, NODE_K_C, NHID_C, NHID_C/16,
            nw1s + (size_t)l * NODE_KP_C * NHID_C);
        swizzle_w_kernel<<<(NHID_C/32)*(NODE_NP_C/16), 32, 0, stream>>>(
            node_w2 + (size_t)l * NHID_C * NF_C, NHID_C, NF_C, NODE_NP_C/16,
            nw2s + (size_t)l * NHID_C * NODE_NP_C);
    }

    // Embed + edge features + graph counts
    embed_kernel<<<(N_NODES_C + 255) / 256, 256, 0, stream>>>(
        x, embs[0], embs[1], embs[2], embs[3], embs[4], embs[5], embs[6],
        embs[7], embs[8], embs[9], feats, feats16);
    ea_kernel<<<(EPRIME_C + 255) / 256, 256, 0, stream>>>(ei, eattr, pos, ea16);
    hipMemsetAsync(counts, 0, N_GRAPHS_C * 4, stream);
    count_kernel<<<(N_NODES_C + 255) / 256, 256, 0, stream>>>(batch, counts);

    for (int l = 0; l < 2; ++l) {
        hipMemsetAsync(m_i, 0, (size_t)N_NODES_C * MDIM_C * 4, stream);
        hipMemsetAsync(gsum, 0, 2 * N_GRAPHS_C * 4, stream);  // gsum + gsumsq adjacent
        edge_mlp_kernel<<<(EPRIME_C + 63) / 64, 128, 0, stream>>>(
            feats16, ea16, ei,
            w1s + (size_t)l * EF_KP_C * EHID_C, edge_b1 + (size_t)l * EHID_C,
            w2s + (size_t)l * EHID_C * MDIM_C,  edge_b2 + (size_t)l * MDIM_C,
            m_i);
        stats_kernel<<<(N_NODES_C + 255) / 256, 256, 0, stream>>>(feats, batch, gsum, gsumsq);
        lnstat_fin_kernel<<<1, 64, 0, stream>>>(counts, gsum, gsumsq, gmean, ginvstd);
        node_mlp_kernel<<<(N_NODES_C + 63) / 64, 128, 0, stream>>>(
            feats, feats16, m_i, batch, gmean, ginvstd,
            ln_w + (size_t)l * NF_C, ln_b + (size_t)l * NF_C,
            nw1s + (size_t)l * NODE_KP_C * NHID_C, node_b1 + (size_t)l * NHID_C,
            nw2s + (size_t)l * NHID_C * NODE_NP_C, node_b2 + (size_t)l * NF_C);
    }

    hipMemsetAsync(out, 0, (size_t)N_GRAPHS_C * NF_C * 4, stream);
    pool_kernel<<<(N_NODES_C + 255) / 256, 256, 0, stream>>>(feats, batch, out);
    pool_div_kernel<<<(N_GRAPHS_C * NF_C + 255) / 256, 256, 0, stream>>>(out, counts);
}
